// Decoder_2800318677558
// MI455X (gfx1250) — compile-verified, matrix operand order checked by reading, not run on hardware
//
#include <hip/hip_runtime.h>
#include <hip/hip_bf16.h>
#include <math.h>

// ---------------------------------------------------------------------------
// Tacotron2-style decoder for MI455X (gfx1250), persistent-kernel design.
//   B=32, T_IN=256, T_OUT=512, E=512, A_RNN=D_RNN=1024, PRENET=256,
//   A_DIM=128, N_FILT=32, KSIZE=31, N_MEL=80
// LSTM gate GEMMs use v_wmma_f32_16x16x32_bf16 with bf16-converted fused
// weights [wih|whh] resident in L2 (36 MB bf16 << 192 MB L2).
// ---------------------------------------------------------------------------

typedef __attribute__((ext_vector_type(16))) __bf16 v16bf;
typedef __attribute__((ext_vector_type(8)))  float  v8f;

#define NWG   32
#define NTHR  256
#define KATT  1792           // PRENET+E (768) + A_RNN (1024)
#define KDEC  2560           // A_RNN+E (1536) + D_RNN (1024)

__device__ __forceinline__ unsigned short f2bf(float x) {
  unsigned u = __float_as_uint(x);
  unsigned r = u + 0x7FFFu + ((u >> 16) & 1u);
  return (unsigned short)(r >> 16);
}
__device__ __forceinline__ float sigm(float x) { return 1.0f / (1.0f + __expf(-x)); }

// Load one 16-element bf16 fragment (two 16B chunks: k..k+7 and k+16..k+23)
__device__ __forceinline__ v16bf ld_frag(const unsigned short* base) {
  union { v16bf v; uint4 u[2]; } r;
  r.u[0] = *(const uint4*)(base);
  r.u[1] = *(const uint4*)(base + 16);
  return r.v;
}

// ---------------------------------------------------------------------------
// Grid barrier (32 co-resident workgroups)
// ---------------------------------------------------------------------------
__device__ __forceinline__ void grid_sync(unsigned* cnt, volatile unsigned* gen) {
  __syncthreads();
  if (threadIdx.x == 0) {
    __threadfence();
    unsigned g = *gen;
    if (atomicAdd(cnt, 1u) == NWG - 1u) {
      *cnt = 0u;
      __threadfence();
      *gen = g + 1u;
    } else {
      while (*gen == g) __builtin_amdgcn_s_sleep(2);
    }
    __threadfence();
  }
  __syncthreads();
}

// ---------------------------------------------------------------------------
// One wave computes one 32x16 tile of gates = A[32xK](bf16) * W^T[Kx16](bf16)
// gwave in [0,256): N-tile index. W stored row-major (4096 x K) bf16.
// ---------------------------------------------------------------------------
__device__ void gemm_gates(const unsigned short* A, const unsigned short* W,
                           float* C, int K, int gwave, int lane) {
  const int n0   = gwave << 4;
  const int lrow = lane & 15;
  const int hi   = (lane >> 4) << 3;   // lanes 16-31 take K chunk +8
  v8f acc0 = {0.f,0.f,0.f,0.f,0.f,0.f,0.f,0.f};
  v8f acc1 = {0.f,0.f,0.f,0.f,0.f,0.f,0.f,0.f};
  const unsigned short* wp = W + (size_t)(n0 + lrow) * K + hi;
  const unsigned short* a0 = A + (size_t)lrow * K + hi;
  const unsigned short* a1 = A + (size_t)(lrow + 16) * K + hi;
  for (int kb = 0; kb < K; kb += 32) {
    v16bf bt  = ld_frag(wp + kb);
    v16bf at0 = ld_frag(a0 + kb);
    v16bf at1 = ld_frag(a1 + kb);
    acc0 = __builtin_amdgcn_wmma_f32_16x16x32_bf16(false, at0, false, bt,
                                                   (short)0, acc0, false, false);
    acc1 = __builtin_amdgcn_wmma_f32_16x16x32_bf16(false, at1, false, bt,
                                                   (short)0, acc1, false, false);
  }
  const int col = n0 + lrow;
  const int mb  = (lane >> 4) << 3;
#pragma unroll
  for (int j = 0; j < 8; ++j) {
    C[(size_t)(mb + j)      * 4096 + col] = acc0[j];
    C[(size_t)(mb + j + 16) * 4096 + col] = acc1[j];
  }
}

// ---------------------------------------------------------------------------
// Setup kernels
// ---------------------------------------------------------------------------
__global__ void k_zero(unsigned* p, size_t n) {
  for (size_t i = blockIdx.x * (size_t)blockDim.x + threadIdx.x; i < n;
       i += (size_t)gridDim.x * blockDim.x)
    p[i] = 0u;
}

// Fuse + convert LSTM weights to bf16:  WA[n][k] = k<768 ? wih : whh
__global__ void k_convert(const float* awih, const float* awhh,
                          const float* dwih, const float* dwhh,
                          unsigned short* WA, unsigned short* WD) {
  const size_t na = (size_t)4096 * KATT;
  const size_t nd = (size_t)4096 * KDEC;
  for (size_t i = blockIdx.x * (size_t)blockDim.x + threadIdx.x; i < na + nd;
       i += (size_t)gridDim.x * blockDim.x) {
    if (i < na) {
      int n = (int)(i / KATT), k = (int)(i % KATT);
      float v = (k < 768) ? awih[(size_t)n * 768 + k] : awhh[(size_t)n * 1024 + (k - 768)];
      WA[i] = f2bf(v);
    } else {
      size_t j = i - na;
      int n = (int)(j / KDEC), k = (int)(j % KDEC);
      float v = (k < 1536) ? dwih[(size_t)n * 1536 + k] : dwhh[(size_t)n * 1024 + (k - 1536)];
      WD[j] = f2bf(v);
    }
  }
}

// Prenet for every teacher-forced step (t=0 is the zero go-frame).
// out: bf16 (T_OUT, B, 256)
__global__ void k_prenet(const float* di, const float* w1, const float* w2,
                         unsigned short* out) {
  const int t = blockIdx.x >> 5;
  const int b = blockIdx.x & 31;
  __shared__ float fr[80];
  __shared__ float x1[256];
  const int tid = threadIdx.x;
  if (tid < 80) fr[tid] = (t == 0) ? 0.f : di[((size_t)b * 80 + tid) * 512 + (t - 1)];
  __syncthreads();
  float a = 0.f;
  for (int m = 0; m < 80; ++m) a += fr[m] * w1[tid * 80 + m];
  x1[tid] = fmaxf(a, 0.f);
  __syncthreads();
  float c = 0.f;
  for (int q = 0; q < 256; ++q) c += x1[q] * w2[tid * 256 + q];
  out[((size_t)t * 32 + b) * 256 + tid] = f2bf(fmaxf(c, 0.f));
}

// processed_memory (B*T_IN, 128) = memory (B*T_IN, 512) @ wm.T
__global__ void k_pm(const float* mem, const float* wm, float* pm) {
  const int bt = blockIdx.x;          // b*256 + t
  __shared__ float row[512];
  const int tid = threadIdx.x;        // 128
  for (int i = tid; i < 512; i += 128) row[i] = mem[(size_t)bt * 512 + i];
  __syncthreads();
  float a = 0.f;
  for (int k = 0; k < 512; ++k) a += row[k] * wm[tid * 512 + k];
  pm[(size_t)bt * 128 + tid] = a;
}

// ---------------------------------------------------------------------------
// Persistent decoder kernel
// ---------------------------------------------------------------------------
struct P {
  // inputs
  const float *memory, *arb, *wq, *vv, *lconv, *ldense, *drb;
  const float *projw, *projb, *gatew, *gateb;
  const int   *memlen;
  // workspace
  const unsigned short *WA, *WD, *pre_bf;
  unsigned short *stA, *stD;           // bf16 staging: [pre|ctx|ah], [ah|ctx|dh]
  float *pm, *gA, *gD, *pqv, *ener;
  float *ah, *ac, *dh, *dc, *ctx, *aw, *awc;
  // outputs
  float *out_mel, *out_gate, *out_align;
  unsigned *bcnt; volatile unsigned *bgen;
};

__global__ __launch_bounds__(NTHR) void k_decoder(P p) {
  const int tid   = threadIdx.x;
  const int gid   = blockIdx.x * NTHR + tid;     // 0..8191
  const int lane  = tid & 31;
  const int gwave = (blockIdx.x << 3) + (tid >> 5);  // 0..255
  __shared__ float sh[512];

  for (int t = 0; t < 512; ++t) {
    grid_sync(p.bcnt, p.bgen);                 // staging [pre_t | ctx | ah] ready

    // ---- attention LSTM gates: (32 x 4096) = stA (32x1792) @ WA^T ----
    gemm_gates(p.stA, p.WA, p.gA, KATT, gwave, lane);
    grid_sync(p.bcnt, p.bgen);

    // ---- attention LSTM cell ----
    for (int i = gid; i < 32768; i += NWG * NTHR) {
      int b = i >> 10, n = i & 1023;
      const float* g = p.gA + (size_t)b * 4096;
      float gi = g[n]        + p.arb[n];
      float gf = g[1024 + n] + p.arb[1024 + n];
      float gg = g[2048 + n] + p.arb[2048 + n];
      float go = g[3072 + n] + p.arb[3072 + n];
      float c2 = sigm(gf) * p.ac[i] + sigm(gi) * tanhf(gg);
      float h2 = sigm(go) * tanhf(c2);
      p.ac[i] = c2; p.ah[i] = h2;
      unsigned short hb = f2bf(h2);
      p.stA[(size_t)b * KATT + 768 + n] = hb;   // for next step's att GEMM
      p.stD[(size_t)b * KDEC + n]       = hb;   // for this step's dec GEMM
    }
    grid_sync(p.bcnt, p.bgen);

    // ---- query projection pq = ah @ wq.T  (32x128) ----
    if (gid < 4096) {
      int b = gid >> 7, d = gid & 127;
      const float* hr = p.ah + (size_t)b * 1024;
      const float* wr = p.wq + (size_t)d * 1024;
      float a = 0.f;
      for (int k = 0; k < 1024; ++k) a += hr[k] * wr[k];
      p.pqv[gid] = a;
    }
    grid_sync(p.bcnt, p.bgen);

    // ---- location conv + energies: one thread per (b, t_in) ----
    {
      int b = gid >> 8, ti = gid & 255;
      float conv[32];
#pragma unroll
      for (int f = 0; f < 32; ++f) conv[f] = 0.f;
      const float* awr  = p.aw  + b * 256;
      const float* awcr = p.awc + b * 256;
      for (int j = 0; j < 31; ++j) {
        int tt = ti + j - 15;
        if (tt < 0 || tt >= 256) continue;
        float a0 = awr[tt], a1 = awcr[tt];
        for (int f = 0; f < 32; ++f)
          conv[f] += p.lconv[f * 62 + j] * a0 + p.lconv[f * 62 + 31 + j] * a1;
      }
      const float* pmr = p.pm  + (size_t)gid * 128;
      const float* pqr = p.pqv + b * 128;
      float e = 0.f;
      for (int d = 0; d < 128; ++d) {
        float s = pqr[d] + pmr[d];
        const float* ldr = p.ldense + d * 32;
        for (int f = 0; f < 32; ++f) s += ldr[f] * conv[f];
        e += p.vv[d] * tanhf(s);
      }
      if (ti >= p.memlen[b]) e = -1e9f;
      p.ener[gid] = e;
    }
    grid_sync(p.bcnt, p.bgen);

    // ---- softmax + context: workgroup b handles batch row b ----
    {
      const int b = blockIdx.x;
      float e = p.ener[b * 256 + tid];
      sh[tid] = e; __syncthreads();
      for (int s = 128; s > 0; s >>= 1) {
        if (tid < s) sh[tid] = fmaxf(sh[tid], sh[tid + s]);
        __syncthreads();
      }
      float mx = sh[0]; __syncthreads();
      float ex = __expf(e - mx);
      sh[tid] = ex; __syncthreads();
      for (int s = 128; s > 0; s >>= 1) {
        if (tid < s) sh[tid] += sh[tid + s];
        __syncthreads();
      }
      float w = ex / sh[0]; __syncthreads();
      sh[tid] = w;                           // weights for ctx
      p.aw[b * 256 + tid]  = w;
      p.awc[b * 256 + tid] += w;
      p.out_align[((size_t)b * 512 + t) * 256 + tid] = w;
      __syncthreads();
      for (int e2 = tid; e2 < 512; e2 += 256) {
        float a = 0.f;
        const float* mr = p.memory + (size_t)b * 256 * 512 + e2;
        for (int tt = 0; tt < 256; ++tt) a += sh[tt] * mr[(size_t)tt * 512];
        p.ctx[b * 512 + e2] = a;
        unsigned short cb = f2bf(a);
        p.stD[(size_t)b * KDEC + 1024 + e2] = cb;
        p.stA[(size_t)b * KATT + 256 + e2]  = cb;   // next step's att input
      }
    }
    grid_sync(p.bcnt, p.bgen);

    // ---- decoder LSTM gates: (32 x 4096) = stD (32x2560) @ WD^T ----
    gemm_gates(p.stD, p.WD, p.gD, KDEC, gwave, lane);
    grid_sync(p.bcnt, p.bgen);

    // ---- decoder LSTM cell ----
    for (int i = gid; i < 32768; i += NWG * NTHR) {
      int b = i >> 10, n = i & 1023;
      const float* g = p.gD + (size_t)b * 4096;
      float gi = g[n]        + p.drb[n];
      float gf = g[1024 + n] + p.drb[1024 + n];
      float gg = g[2048 + n] + p.drb[2048 + n];
      float go = g[3072 + n] + p.drb[3072 + n];
      float c2 = sigm(gf) * p.dc[i] + sigm(gi) * tanhf(gg);
      float h2 = sigm(go) * tanhf(c2);
      p.dc[i] = c2; p.dh[i] = h2;
      p.stD[(size_t)b * KDEC + 1536 + n] = f2bf(h2);  // next step's dec GEMM
    }
    grid_sync(p.bcnt, p.bgen);

    // ---- projections + outputs + stage next prenet frame ----
    {
      if (gid < 2560) {
        int b = gid / 80, m = gid % 80;
        const float* dhr = p.dh  + (size_t)b * 1024;
        const float* cr  = p.ctx + b * 512;
        const float* wr  = p.projw + (size_t)m * 1536;
        float a = p.projb[m];
        for (int k = 0; k < 1024; ++k) a += dhr[k] * wr[k];
        for (int k = 0; k < 512;  ++k) a += cr[k]  * wr[1024 + k];
        p.out_mel[((size_t)b * 80 + m) * 512 + t] = a;
      } else if (gid < 2592) {
        int b = gid - 2560;
        const float* dhr = p.dh  + (size_t)b * 1024;
        const float* cr  = p.ctx + b * 512;
        float a = p.gateb[0];
        for (int k = 0; k < 1024; ++k) a += dhr[k] * p.gatew[k];
        for (int k = 0; k < 512;  ++k) a += cr[k]  * p.gatew[1024 + k];
        p.out_gate[(size_t)b * 512 + t] = a;
      }
      if (t + 1 < 512) {
        int b = gid >> 8, pp = gid & 255;
        p.stA[(size_t)b * KATT + pp] =
            p.pre_bf[(((size_t)(t + 1)) * 32 + b) * 256 + pp];
      }
    }
  }
}

// ---------------------------------------------------------------------------
// Host launcher
// ---------------------------------------------------------------------------
extern "C" void kernel_launch(void* const* d_in, const int* in_sizes, int n_in,
                              void* d_out, int out_size, void* d_ws, size_t ws_size,
                              hipStream_t stream) {
  const float* memory = (const float*)d_in[0];
  const float* dec_in = (const float*)d_in[1];
  const int*   memlen = (const int*)  d_in[2];
  const float* pw1    = (const float*)d_in[3];
  const float* pw2    = (const float*)d_in[4];
  const float* awih   = (const float*)d_in[5];
  const float* awhh   = (const float*)d_in[6];
  const float* arb    = (const float*)d_in[7];
  const float* wq     = (const float*)d_in[8];
  const float* wm     = (const float*)d_in[9];
  const float* vv     = (const float*)d_in[10];
  const float* lconv  = (const float*)d_in[11];
  const float* ldense = (const float*)d_in[12];
  const float* dwih   = (const float*)d_in[13];
  const float* dwhh   = (const float*)d_in[14];
  const float* drb    = (const float*)d_in[15];
  const float* projw  = (const float*)d_in[16];
  const float* projb  = (const float*)d_in[17];
  const float* gatew  = (const float*)d_in[18];
  const float* gateb  = (const float*)d_in[19];
  (void)in_sizes; (void)n_in; (void)out_size; (void)ws_size;

  char* cur = (char*)d_ws;
  auto take = [&](size_t bytes) {
    char* r = cur; cur += (bytes + 255) & ~(size_t)255; return r;
  };

  unsigned short* WA  = (unsigned short*)take((size_t)4096 * KATT * 2);
  unsigned short* WD  = (unsigned short*)take((size_t)4096 * KDEC * 2);
  unsigned short* pre = (unsigned short*)take((size_t)512 * 32 * 256 * 2);
  float* pm   = (float*)take((size_t)32 * 256 * 128 * 4);
  float* gA   = (float*)take((size_t)32 * 4096 * 4);
  float* gD   = (float*)take((size_t)32 * 4096 * 4);
  float* pqv  = (float*)take((size_t)32 * 128 * 4);
  float* ener = (float*)take((size_t)32 * 256 * 4);
  // zero-initialized state region (contiguous)
  char* z0 = cur;
  float* ah  = (float*)take((size_t)32 * 1024 * 4);
  float* ac  = (float*)take((size_t)32 * 1024 * 4);
  float* dh  = (float*)take((size_t)32 * 1024 * 4);
  float* dc  = (float*)take((size_t)32 * 1024 * 4);
  float* ctx = (float*)take((size_t)32 * 512 * 4);
  float* aw  = (float*)take((size_t)32 * 256 * 4);
  float* awc = (float*)take((size_t)32 * 256 * 4);
  unsigned short* stA = (unsigned short*)take((size_t)32 * KATT * 2);
  unsigned short* stD = (unsigned short*)take((size_t)32 * KDEC * 2);
  unsigned* bar = (unsigned*)take(2 * sizeof(unsigned));
  size_t zbytes = (size_t)(cur - z0);

  float* out_mel   = (float*)d_out;
  float* out_gate  = out_mel + (size_t)32 * 80 * 512;
  float* out_align = out_gate + (size_t)32 * 512;

  k_zero   <<<256, 256, 0, stream>>>((unsigned*)z0, zbytes / 4);
  k_convert<<<512, 256, 0, stream>>>(awih, awhh, dwih, dwhh, WA, WD);
  k_prenet <<<512 * 32, 256, 0, stream>>>(dec_in, pw1, pw2, pre);
  k_pm     <<<8192, 128, 0, stream>>>(memory, wm, pm);

  P p;
  p.memory = memory; p.arb = arb; p.wq = wq; p.vv = vv;
  p.lconv = lconv; p.ldense = ldense; p.drb = drb;
  p.projw = projw; p.projb = projb; p.gatew = gatew; p.gateb = gateb;
  p.memlen = memlen;
  p.WA = WA; p.WD = WD; p.pre_bf = pre;
  p.stA = stA; p.stD = stD;
  p.pm = pm; p.gA = gA; p.gD = gD; p.pqv = pqv; p.ener = ener;
  p.ah = ah; p.ac = ac; p.dh = dh; p.dc = dc;
  p.ctx = ctx; p.aw = aw; p.awc = awc;
  p.out_mel = out_mel; p.out_gate = out_gate; p.out_align = out_align;
  p.bcnt = bar; p.bgen = bar + 1;

  k_decoder<<<NWG, NTHR, 0, stream>>>(p);
}
